// NeuralControlCritic_39049842655977
// MI455X (gfx1250) — matrix operand order repaired and checked by reading, not compiled, not run on hardware
//
#include <hip/hip_runtime.h>
#include <hip/hip_bf16.h>

// ---------------- model dims ----------------
#define STATE_SIZE 2048
#define SEQ        256
#define HID        512
#define D_INNER    1024
#define D_STATE    128
#define D_CONV     4
#define HEADDIM    64
#define NHEADS     16
#define CONV_DIM   1280           // D_INNER + 2*D_STATE
#define D_IN_PROJ  2320           // 2*D_INNER + 2*D_STATE + NHEADS
#define NBLK       4

// ---------------- WMMA / TDM types ----------------
typedef __attribute__((ext_vector_type(16))) __bf16          v16bf;
typedef __attribute__((ext_vector_type(8)))  float           v8f;
typedef __attribute__((ext_vector_type(8)))  unsigned short  ushort8;
typedef __attribute__((ext_vector_type(4)))  unsigned int    uint4v;
typedef __attribute__((ext_vector_type(8)))  int             int8v;
typedef __attribute__((ext_vector_type(4)))  int             int4v;

union BFrag {                  // 32 bytes = 8 VGPRs = one 16-bit WMMA operand
    v16bf          v;
    ushort8        h[2];
    unsigned short u[16];
};

__device__ __forceinline__ unsigned short f2bf(float f) {
    unsigned int u = __builtin_bit_cast(unsigned int, f);
    unsigned int r = u + 0x7FFFu + ((u >> 16) & 1u);   // round-to-nearest-even
    return (unsigned short)(r >> 16);
}

__device__ __forceinline__ float siluf(float x) { return x / (1.f + expf(-x)); }

// ---- Tensor Data Mover: 2D bf16 tile (tile_rows x tile_k elems) global -> LDS ----
// D# packing per CDNA5 ISA 8.3/8.4: group0 = {flags, lds_addr, global_addr, type=2},
// group1 = {data_size=2B, tensor_dim0/1, tile_dim0/1, dim0 stride}. 2D: groups 2/3 zero.
__device__ __forceinline__ void tdm_load_tile_bf16(unsigned lds_off,
                                                   const unsigned short* gptr,
                                                   unsigned tensor_k, unsigned tensor_rows,
                                                   unsigned tile_k, unsigned tile_rows,
                                                   unsigned row_stride_elems) {
    unsigned long long ga = (unsigned long long)(const void*)gptr;
    uint4v g0;
    g0[0] = 1u;                                            // count=1 (valid user D#)
    g0[1] = lds_off;                                       // lds_addr (bytes)
    g0[2] = (unsigned)(ga & 0xFFFFFFFFu);                  // global_addr[31:0]
    g0[3] = (unsigned)((ga >> 32) & 0x01FFFFFFu) | (2u << 30);  // addr[56:32] | type=2
    int8v g1;
    g1[0] = (int)(1u << 16);                               // data_size=1 -> 2 bytes
    g1[1] = (int)((tensor_k & 0xFFFFu) << 16);             // tensor_dim0[15:0]
    g1[2] = (int)(((tensor_k >> 16) & 0xFFFFu) | ((tensor_rows & 0xFFFFu) << 16));
    g1[3] = (int)(((tensor_rows >> 16) & 0xFFFFu) | ((tile_k & 0xFFFFu) << 16)); // tile_dim0
    g1[4] = (int)(tile_rows & 0xFFFFu);                    // tile_dim1 (tile_dim2=0)
    g1[5] = (int)row_stride_elems;                         // tensor_dim0_stride[31:0]
    g1[6] = 0;                                             // stride[47:32], dim1_stride lo
    g1[7] = 0;
    int4v gz = {0, 0, 0, 0};
#if __clang_major__ >= 23
    int8v gz8 = {0, 0, 0, 0, 0, 0, 0, 0};
    __builtin_amdgcn_tensor_load_to_lds(g0, g1, gz, gz, gz8, 0);
#else
    __builtin_amdgcn_tensor_load_to_lds(g0, g1, gz, gz, 0);
#endif
}

// ---------------- fp32 -> bf16 convert ----------------
__global__ void k_cvt_bf16(const float* __restrict__ src, unsigned short* __restrict__ dst, int n) {
    int i = blockIdx.x * blockDim.x + threadIdx.x;
    if (i < n) dst[i] = f2bf(src[i]);
}

// ---------------- GEMV: warp per row, K % 128 == 0 ----------------
__global__ void k_gemv_w(const float* __restrict__ W, const float* __restrict__ x,
                         const float* __restrict__ b, float* __restrict__ y,
                         int rows, int K, int relu) {
    int w = blockIdx.x * (blockDim.x >> 5) + (threadIdx.x >> 5);
    if (w >= rows) return;
    int lane = threadIdx.x & 31;
    const float* row = W + (size_t)w * K;
    float s = 0.f;
    for (int k = 4 * lane; k < K; k += 128) {
        float4 wv = *(const float4*)(row + k);
        float4 xv = *(const float4*)(x + k);
        s += wv.x * xv.x + wv.y * xv.y + wv.z * xv.z + wv.w * xv.w;
    }
    for (int m = 16; m >= 1; m >>= 1) s += __shfl_xor(s, m, 32);
    if (lane == 0) {
        if (b) s += b[w];
        if (relu) s = fmaxf(s, 0.f);
        y[w] = s;
    }
}

// ---------------- GEMV: one 256-thread block per row (huge K) ----------------
__global__ void k_gemv_b(const float* __restrict__ W, const float* __restrict__ x,
                         const float* __restrict__ b, float* __restrict__ y,
                         int K, int relu) {
    __shared__ float red[256];
    int r = blockIdx.x;
    const float* row = W + (size_t)r * K;
    float s = 0.f;
    for (int k = 4 * threadIdx.x; k < K; k += 1024) {
        __builtin_prefetch(row + k + 4096, 0, 0);      // global_prefetch_b8
        float4 wv = *(const float4*)(row + k);
        float4 xv = *(const float4*)(x + k);
        s += wv.x * xv.x + wv.y * xv.y + wv.z * xv.z + wv.w * xv.w;
    }
    red[threadIdx.x] = s;
    __syncthreads();
    for (int m = 128; m >= 1; m >>= 1) {
        if ((int)threadIdx.x < m) red[threadIdx.x] += red[threadIdx.x + m];
        __syncthreads();
    }
    if (threadIdx.x == 0) {
        s = red[0];
        if (b) s += b[r];
        if (relu) s = fmaxf(s, 0.f);
        y[r] = s;
    }
}

// ---------------- input layer 1: xc(2050) @ W1.T, relu ----------------
__global__ void k_in1(const float* __restrict__ W, const float* __restrict__ x,
                      const float* __restrict__ steps, const float* __restrict__ cur,
                      const float* __restrict__ b, float* __restrict__ out) {
    int w = blockIdx.x * (blockDim.x >> 5) + (threadIdx.x >> 5);
    if (w >= HID) return;
    int lane = threadIdx.x & 31;
    const float* row = W + (size_t)w * (STATE_SIZE + 2);   // rows are 8B aligned
    float s = 0.f;
    for (int k = 2 * lane; k < STATE_SIZE; k += 64) {
        float2 wv = *(const float2*)(row + k);
        float2 xv = *(const float2*)(x + k);
        s += wv.x * xv.x + wv.y * xv.y;
    }
    if (lane == 0) s += row[STATE_SIZE] * steps[0] + row[STATE_SIZE + 1] * cur[0];
    for (int m = 16; m >= 1; m >>= 1) s += __shfl_xor(s, m, 32);
    if (lane == 0) out[w] = fmaxf(s + b[w], 0.f);
}

// ======== WMMA fragment helpers (CDNA5 ISA 7.12.2 wave32 layouts) ========
//   A (16x32 bf16): lane L -> row M=L%16; elems 0..7 -> K = 8*(L/16)+e,
//                   elems 8..15 -> K = 16 + 8*(L/16) + (e-8)
//   B (32x16 bf16): lane L -> col N=L%16; same K mapping as A
//   C/D (f32):      VGPR j -> row M = j + 8*(L/16), col N = L%16
__device__ __forceinline__ v16bf frag_a(const unsigned short* __restrict__ arow, int hi) {
    BFrag a;
    a.h[0] = *(const ushort8*)(arow + hi * 8);
    a.h[1] = *(const ushort8*)(arow + 16 + hi * 8);
    return a.v;
}
__device__ __forceinline__ v16bf frag_b(const unsigned short* __restrict__ brow, int hi) {
    BFrag b;
    b.h[0] = *(const ushort8*)(brow + hi * 8);
    b.h[1] = *(const ushort8*)(brow + 16 + hi * 8);
    return b.v;
}

// ---------------- bf16 WMMA GEMM (direct loads): C[M,N] = A[M,K] * W[N,K]^T ----------------
__global__ void k_gemm_bf16(const unsigned short* __restrict__ A,  // [M,K] bf16
                            const unsigned short* __restrict__ B,  // [N,K] bf16
                            float* __restrict__ C,                 // [M,N] f32
                            int M, int N, int K) {
    const int lane  = threadIdx.x & 31;
    const int wave  = blockIdx.x * (blockDim.x >> 5) + (threadIdx.x >> 5);
    const int ntile = N >> 4;
    const int mt = wave / ntile;
    const int nt = wave % ntile;
    if (mt * 16 >= M) return;                  // wave-uniform: EXEC stays all-ones
    const int r  = lane & 15;
    const int hi = lane >> 4;

    const unsigned short* arow = A + (size_t)(mt * 16 + r) * K;
    const unsigned short* brow = B + (size_t)(nt * 16 + r) * K;

    v8f acc = {};
    for (int kk = 0; kk < K; kk += 32) {
        acc = __builtin_amdgcn_wmma_f32_16x16x32_bf16(false, frag_a(arow + kk, hi),
                                                      false, frag_b(brow + kk, hi),
                                                      (short)0, acc, false, false);
    }
    float* crow = C + (size_t)(mt * 16 + 8 * hi) * N + nt * 16 + r;
#pragma unroll
    for (int j = 0; j < 8; ++j) crow[(size_t)j * N] = acc[j];
}

// ---------------- bf16 WMMA GEMM, TDM-staged + double-buffered LDS ----------------
// Macro-tile 32(M) x 64(N) per 256-thread block (8 waves = 2x4 of 16x16 tiles).
// Wave 0 drives the Tensor Data Mover: A panel (32x32) + B panel (64x32) per K step,
// ping-pong buffered so the next panels stream in while all waves run WMMA from LDS.
__global__ void k_gemm_bf16_tdm(const unsigned short* __restrict__ A,  // [M,K] bf16
                                const unsigned short* __restrict__ B,  // [N,K] bf16
                                float* __restrict__ C,                 // [M,N] f32
                                int M, int N, int K) {
    __shared__ unsigned short sA[2][32 * 32];   // 2 x 2KB
    __shared__ unsigned short sB[2][64 * 32];   // 2 x 4KB

    const int mblk = M >> 5;
    const int bm = blockIdx.x % mblk;
    const int bn = blockIdx.x / mblk;
    const int wv = threadIdx.x >> 5;
    const int wm = wv >> 2;                 // 0..1
    const int wn = wv & 3;                  // 0..3
    const int lane = threadIdx.x & 31;
    const int r  = lane & 15;
    const int hi = lane >> 4;

    const unsigned short* Ap = A + (size_t)(bm * 32) * K;   // panel base
    const unsigned short* Bp = B + (size_t)(bn * 64) * K;

    const bool driver = (wv == 0);
    if (driver) {
        tdm_load_tile_bf16((unsigned)(size_t)&sA[0][0], Ap, K, M, 32, 32, K);
        tdm_load_tile_bf16((unsigned)(size_t)&sB[0][0], Bp, K, N, 32, 64, K);
    }

    v8f acc = {};
    for (int kk = 0; kk < K; kk += 32) {
        const int cur = (kk >> 5) & 1;
        if (driver) __builtin_amdgcn_s_wait_tensorcnt(0);   // cur panels landed
        __syncthreads();                                    // visible to all waves
        if (driver && (kk + 32) < K) {                      // prefetch next panels
            tdm_load_tile_bf16((unsigned)(size_t)&sA[cur ^ 1][0], Ap + (kk + 32), K, M, 32, 32, K);
            tdm_load_tile_bf16((unsigned)(size_t)&sB[cur ^ 1][0], Bp + (kk + 32), K, N, 32, 64, K);
        }
        const unsigned short* arow = &sA[cur][(wm * 16 + r) * 32];  // ds_load_b128
        const unsigned short* brow = &sB[cur][(wn * 16 + r) * 32];
        acc = __builtin_amdgcn_wmma_f32_16x16x32_bf16(false, frag_a(arow, hi),
                                                      false, frag_b(brow, hi),
                                                      (short)0, acc, false, false);
        __syncthreads();                                    // done reading cur
    }
    float* crow = C + (size_t)(bm * 32 + wm * 16 + 8 * hi) * N + bn * 64 + wn * 16 + r;
#pragma unroll
    for (int j = 0; j < 8; ++j) crow[(size_t)j * N] = acc[j];
}

// ---------------- dt = softplus(raw + bias), dA = exp(dt * -exp(A_log)) ----------------
__global__ void k_dt(const float* __restrict__ zx, const float* __restrict__ dt_bias,
                     const float* __restrict__ A_log,
                     float* __restrict__ dt, float* __restrict__ dA) {
    int gid = blockIdx.x * blockDim.x + threadIdx.x;
    if (gid >= SEQ * NHEADS) return;
    int s = gid >> 4, h = gid & 15;
    float raw = zx[(size_t)s * D_IN_PROJ + (D_IN_PROJ - NHEADS) + h] + dt_bias[h];
    float d = raw > 20.f ? raw : log1pf(expf(raw));
    float A = -expf(A_log[h]);
    dt[gid] = d;
    dA[gid] = expf(d * A);
}

// ---------------- depthwise causal conv (width 4) + silu ----------------
__global__ void k_conv(const float* __restrict__ zx, const float* __restrict__ cw,
                       const float* __restrict__ cb, float* __restrict__ xBC) {
    int gid = blockIdx.x * blockDim.x + threadIdx.x;
    if (gid >= SEQ * CONV_DIM) return;
    int s = gid / CONV_DIM, c = gid % CONV_DIM;
    float acc = cb[c];
#pragma unroll
    for (int k = 0; k < D_CONV; ++k) {
        int sp = s - (D_CONV - 1) + k;
        if (sp >= 0) acc += zx[(size_t)sp * D_IN_PROJ + D_INNER + c] * cw[k * CONV_DIM + c];
    }
    xBC[gid] = siluf(acc);
}

// ---------------- selective-scan: one block per head, state in registers ----------------
__global__ void k_scan(const float* __restrict__ xBC, const float* __restrict__ dt,
                       const float* __restrict__ dA, const float* __restrict__ Dh,
                       float* __restrict__ y) {
    const int head = blockIdx.x;
    const int tid  = threadIdx.x;
    const int p = tid >> 2;            // head-dim row 0..63
    const int q = tid & 3;             // owns n in [q*32, q*32+32)
    __shared__ float Bsh[D_STATE];
    __shared__ float Csh[D_STATE];
    float hs[32];
#pragma unroll
    for (int j = 0; j < 32; ++j) hs[j] = 0.f;
    const float Dv = Dh[head];

    for (int t = 0; t < SEQ; ++t) {
        __syncthreads();
        if (tid < 128) Bsh[tid]       = xBC[(size_t)t * CONV_DIM + D_INNER + tid];
        else           Csh[tid - 128] = xBC[(size_t)t * CONV_DIM + D_INNER + D_STATE + (tid - 128)];
        __syncthreads();
        const float xv   = xBC[(size_t)t * CONV_DIM + head * HEADDIM + p];
        const float dtv  = dt[t * NHEADS + head];
        const float dav  = dA[t * NHEADS + head];
        const float coef = dtv * xv;
        const float* Bq = Bsh + q * 32;
        const float* Cq = Csh + q * 32;
        float acc = 0.f;
#pragma unroll
        for (int j = 0; j < 32; ++j) {
            hs[j] = dav * hs[j] + coef * Bq[j];
            acc  += hs[j] * Cq[j];
        }
        acc += __shfl_xor(acc, 1, 32);     // reduce across the 4 lanes sharing p
        acc += __shfl_xor(acc, 2, 32);
        if (q == 0) y[(size_t)t * D_INNER + head * HEADDIM + p] = acc + Dv * xv;
    }
}

// ---------------- gated RMSNorm -> bf16 (GEMM A operand) ----------------
__global__ void k_rmsnorm(const float* __restrict__ y, const float* __restrict__ zx,
                          const float* __restrict__ nw, unsigned short* __restrict__ outbf) {
    __shared__ float red[256];
    __shared__ float gsh[D_INNER];
    int s = blockIdx.x, tid = threadIdx.x;
    float ss = 0.f;
#pragma unroll
    for (int i = 0; i < 4; ++i) {
        int c = tid + i * 256;
        float z = zx[(size_t)s * D_IN_PROJ + c];
        float g = y[(size_t)s * D_INNER + c] * siluf(z);
        gsh[c] = g;
        ss += g * g;
    }
    red[tid] = ss;
    __syncthreads();
    for (int m = 128; m >= 1; m >>= 1) {
        if (tid < m) red[tid] += red[tid + m];
        __syncthreads();
    }
    float scale = rsqrtf(red[0] / (float)D_INNER + 1e-5f);
#pragma unroll
    for (int i = 0; i < 4; ++i) {
        int c = tid + i * 256;
        outbf[(size_t)s * D_INNER + c] = f2bf(gsh[c] * scale * nw[c]);
    }
}

// ============================ host side ============================
extern "C" void kernel_launch(void* const* d_in, const int* in_sizes, int n_in,
                              void* d_out, int out_size, void* d_ws, size_t ws_size,
                              hipStream_t stream) {
    (void)in_sizes; (void)n_in; (void)out_size; (void)ws_size;

    const float* x        = (const float*)d_in[0];
    const float* steps    = (const float*)d_in[1];
    const float* cur      = (const float*)d_in[2];
    const float* mi_W1    = (const float*)d_in[3];
    const float* mi_b1    = (const float*)d_in[4];
    const float* mi_W2    = (const float*)d_in[5];
    const float* mi_b2    = (const float*)d_in[6];
    const float* m_inproj = (const float*)d_in[7];
    const float* m_convw  = (const float*)d_in[8];
    const float* m_convb  = (const float*)d_in[9];
    const float* m_dtb    = (const float*)d_in[10];
    const float* m_Alog   = (const float*)d_in[11];
    const float* m_D      = (const float*)d_in[12];
    const float* m_normw  = (const float*)d_in[13];
    const float* m_outpr  = (const float*)d_in[14];
    const float* mo_W1    = (const float*)d_in[15];
    const float* mo_b1    = (const float*)d_in[16];
    const float* mo_W2    = (const float*)d_in[17];
    const float* mo_b2    = (const float*)d_in[18];
    const float* mo_W3    = (const float*)d_in[19];
    const float* mo_b3    = (const float*)d_in[20];
    const float* mo_W4    = (const float*)d_in[21];
    const float* mo_b4    = (const float*)d_in[22];
    float*       out      = (float*)d_out;

    // ---- workspace layout (256B aligned slabs) ----
    char*  base = (char*)d_ws;
    size_t off  = 0;
    auto  alloc = [&](size_t bytes) { size_t o = off; off += (bytes + 255) & ~(size_t)255; return o; };

    const int nWi = NBLK * D_IN_PROJ * HID;     // 4,751,360
    const int nWo = NBLK * HID * D_INNER;       // 2,097,152
    unsigned short* bfWi = (unsigned short*)(base + alloc((size_t)nWi * 2));
    unsigned short* bfWo = (unsigned short*)(base + alloc((size_t)nWo * 2));
    float* f_h1   = (float*)(base + alloc(HID * 4));
    float* f_u    = (float*)(base + alloc((size_t)SEQ * HID * 4));
    unsigned short* bf_u = (unsigned short*)(base + alloc((size_t)SEQ * HID * 2));
    float* f_zx   = (float*)(base + alloc((size_t)SEQ * D_IN_PROJ * 4));
    float* f_xBC  = (float*)(base + alloc((size_t)SEQ * CONV_DIM * 4));
    float* f_dt   = (float*)(base + alloc((size_t)SEQ * NHEADS * 4));
    float* f_dA   = (float*)(base + alloc((size_t)SEQ * NHEADS * 4));
    float* f_y    = (float*)(base + alloc((size_t)SEQ * D_INNER * 4));
    unsigned short* bf_y = (unsigned short*)(base + alloc((size_t)SEQ * D_INNER * 2));
    float* f_h2   = (float*)(base + alloc(HID * 4));
    float* f_h3   = (float*)(base + alloc(HID * 4));
    float* f_h4   = (float*)(base + alloc(HID * 4));

    // ---- convert GEMM weights to bf16 (per launch; deterministic) ----
    k_cvt_bf16<<<(nWi + 255) / 256, 256, 0, stream>>>(m_inproj, bfWi, nWi);
    k_cvt_bf16<<<(nWo + 255) / 256, 256, 0, stream>>>(m_outpr, bfWo, nWo);

    // ---- input MLP ----
    k_in1<<<HID / 8, 256, 0, stream>>>(mi_W1, x, steps, cur, mi_b1, f_h1);
    // u = h1 @ mi_W2.T + b2 : 131072 rows, K=512  (bandwidth-bound, warp/row)
    k_gemv_w<<<(SEQ * HID) / 8, 256, 0, stream>>>(mi_W2, f_h1, mi_b2, f_u,
                                                  SEQ * HID, HID, 0);

    // ---- 4 Mamba2 blocks ----
    for (int i = 0; i < NBLK; ++i) {
        const unsigned short* Wi = bfWi + (size_t)i * D_IN_PROJ * HID;
        const unsigned short* Wo = bfWo + (size_t)i * HID * D_INNER;

        k_cvt_bf16<<<(SEQ * HID) / 256, 256, 0, stream>>>(f_u, bf_u, SEQ * HID);

        // zxbcdt = u @ Wi.T : [256,2320]; ragged N -> direct-load WMMA GEMM
        {
            int tiles = (SEQ / 16) * (D_IN_PROJ / 16);
            k_gemm_bf16<<<(tiles + 7) / 8, 256, 0, stream>>>(bf_u, Wi, f_zx,
                                                             SEQ, D_IN_PROJ, HID);
        }
        k_dt<<<(SEQ * NHEADS + 255) / 256, 256, 0, stream>>>(f_zx, m_dtb + i * NHEADS,
                                                             m_Alog + i * NHEADS, f_dt, f_dA);
        k_conv<<<(SEQ * CONV_DIM + 255) / 256, 256, 0, stream>>>(
            f_zx, m_convw + (size_t)i * D_CONV * CONV_DIM, m_convb + (size_t)i * CONV_DIM, f_xBC);
        k_scan<<<NHEADS, 256, 0, stream>>>(f_xBC, f_dt, f_dA, m_D + i * NHEADS, f_y);
        k_rmsnorm<<<SEQ, 256, 0, stream>>>(f_y, f_zx, m_normw + (size_t)i * D_INNER, bf_y);

        // u_next = normed @ Wo.T : [256,1024]x[512,1024]^T, TDM-staged WMMA
        {
            int blocks = (SEQ / 32) * (HID / 64);   // 8 x 8 = 64
            k_gemm_bf16_tdm<<<blocks, 256, 0, stream>>>(bf_y, Wo, f_u,
                                                        SEQ, HID, D_INNER);
        }
    }

    // ---- output MLP ----
    k_gemv_b<<<HID, 256, 0, stream>>>(mo_W1, f_u, mo_b1, f_h2, SEQ * HID, 1);
    k_gemv_w<<<HID / 8, 256, 0, stream>>>(mo_W2, f_h2, mo_b2, f_h3, HID, HID, 1);
    k_gemv_w<<<HID / 8, 256, 0, stream>>>(mo_W3, f_h3, mo_b3, f_h4, HID, HID, 1);
    k_gemv_w<<<1, 256, 0, stream>>>(mo_W4, f_h4, mo_b4, out, 1, HID, 0);
}
